// ComposerModule_52201032516318
// MI455X (gfx1250) — matrix operand order repaired and checked by reading, not compiled, not run on hardware
//
#include <hip/hip_runtime.h>
#include <hip/hip_bf16.h>
#include <cstdint>
#include <cstddef>

// ---------------- model constants ----------------
#define MB 32        // batch
#define MS 512       // seq
#define MH 1024      // hidden
#define MO 16        // operators
#define MOUT 1000    // out dim
#define MLAYERS 4
#define MROWS (MB*MS)   // 16384

typedef __bf16 bf16;
typedef __attribute__((ext_vector_type(16))) __bf16 v16bf;
typedef __attribute__((ext_vector_type(8)))  __bf16 v8bf;
typedef __attribute__((ext_vector_type(8)))  float  v8f;
typedef int v4i_gcc __attribute__((vector_size(16)));   // builtin arg type

union FragAB { v16bf v; v8bf h[2]; };

#if __has_builtin(__builtin_amdgcn_global_load_async_to_lds_b128) && \
    __has_builtin(__builtin_amdgcn_s_wait_asynccnt)
#define HAVE_ASYNC_LDS 1
#else
#define HAVE_ASYNC_LDS 0
#endif

// =====================================================================
// Generic bf16 WMMA GEMM:  C[M,N] = alpha * A[M,K] * B[N,K]^T + bias[N]
// A, B row-major bf16. Template tile config:
//   BM x BN workgroup tile, each wave computes (16*WMT) x (16*WNT).
//   128 threads = 4 waves, arranged (BM/(16*WMT)) x (BN/(16*WNT)).
// Double-buffered LDS tiles, stride-40 padding (conflict-free b128 reads).
// Big config stages tiles with GLOBAL_LOAD_ASYNC_TO_LDS_B128 (ASYNCcnt) —
// zero staging VGPRs. N-guarded config uses masked register staging.
// Requires: M % BM == 0, K % 32 == 0.
// Stores f32 (Cf) and/or bf16 (Cb) for n < n_limit, leading dim ldc.
// =====================================================================
#define LDT 40   // LDS tile row stride in bf16 elements (32 data + 8 pad)

template<int BM, int BN, int WMT, int WNT, bool NGUARD>
__global__ __launch_bounds__(128, 1)
void gemm_bf16_wmma(const bf16* __restrict__ A, int lda,
                    const bf16* __restrict__ B, int ldb,
                    const float* __restrict__ bias,
                    float* __restrict__ Cf, bf16* __restrict__ Cb,
                    int ldc, int M, int N, int K,
                    float alpha, int n_limit)
{
    constexpr int WAVES_M = BM / (16 * WMT);
    constexpr int WAVES_N = BN / (16 * WNT);
    constexpr int NW = WAVES_M * WAVES_N;
    constexpr int NT = NW * 32;
    static_assert(NT == 128, "config must use 4 waves");
    constexpr int ACH = (BM * 2 + NT - 1) / NT;  // A 16-half chunks / thread
    constexpr int BCH = (BN * 2 + NT - 1) / NT;  // B 16-half chunks / thread

    __shared__ bf16 sA[2][BM * LDT];
    __shared__ bf16 sB[2][BN * LDT];

    const int tid  = threadIdx.x;
    const int wave = tid >> 5;
    const int lane = tid & 31;
    const int m0 = blockIdx.y * BM;
    const int n0 = blockIdx.x * BN;
    const int wm = (wave % WAVES_M) * (16 * WMT);
    const int wn = (wave / WAVES_M) * (16 * WNT);

    v8f acc[WMT][WNT];
#pragma unroll
    for (int i = 0; i < WMT; ++i)
#pragma unroll
        for (int j = 0; j < WNT; ++j)
            acc[i][j] = v8f{0.f,0.f,0.f,0.f,0.f,0.f,0.f,0.f};

    const int nk = K >> 5;

    // fragment addressing per CDNA5 ISA layouts
    const int arow = wm + (lane & 15);        // A: M = lane&15
    const int akb  = (lane < 16) ? 0 : 8;     // A: K base 0/8, +16 for hi chunk
    const int bkb  = (lane < 16) ? 0 : 8;     // B: K base 0/8, +16 for hi chunk

    // one K-step of WMMA work from LDS buffer `bufi`
    auto computeStep = [&](int bufi) {
        FragAB fa[WMT];
#pragma unroll
        for (int i = 0; i < WMT; ++i) {
            const bf16* pa = &sA[bufi][(arow + i * 16) * LDT + akb];
            fa[i].h[0] = *(const v8bf*)pa;
            fa[i].h[1] = *(const v8bf*)(pa + 16);
        }
#pragma unroll
        for (int j = 0; j < WNT; ++j) {
            FragAB fb;
            const bf16* pb = &sB[bufi][(wn + j * 16 + (lane & 15)) * LDT + bkb];
            fb.h[0] = *(const v8bf*)pb;
            fb.h[1] = *(const v8bf*)(pb + 16);
#pragma unroll
            for (int i = 0; i < WMT; ++i)
                acc[i][j] = __builtin_amdgcn_wmma_f32_16x16x32_bf16(
                    false, fa[i].v, false, fb.v,
                    (short)0, acc[i][j], false, false);
        }
    };

    int buf = 0;

#if HAVE_ASYNC_LDS
    if constexpr (!NGUARD) {
        // ---- async direct global->LDS staging (no data through VGPRs) ----
        auto issueAsync = [&](int kt, int bufi) {
#pragma unroll
            for (int c = 0; c < ACH; ++c) {
                const int idx = tid + c * NT;
                const int row = idx >> 1, col = (idx & 1) * 16;
                const bf16* g = A + (size_t)(m0 + row) * lda + kt * 32 + col;
                __builtin_amdgcn_global_load_async_to_lds_b128(
                    (v4i_gcc*)g,
                    (v4i_gcc*)&sA[bufi][row * LDT + col],
                    0, 0);
            }
#pragma unroll
            for (int c = 0; c < BCH; ++c) {
                const int idx = tid + c * NT;
                const int row = idx >> 1, col = (idx & 1) * 16;
                const bf16* g = B + (size_t)(n0 + row) * ldb + kt * 32 + col;
                __builtin_amdgcn_global_load_async_to_lds_b128(
                    (v4i_gcc*)g,
                    (v4i_gcc*)&sB[bufi][row * LDT + col],
                    0, 0);
            }
        };

        issueAsync(0, 0);
        __builtin_amdgcn_s_wait_asynccnt(0);
        __syncthreads();
        for (int kt = 0; kt < nk; ++kt) {
            const bool has = (kt + 1) < nk;
            // buf^1 was last read before the previous barrier -> safe target
            if (has) issueAsync(kt + 1, buf ^ 1);
            computeStep(buf);
            if (has) __builtin_amdgcn_s_wait_asynccnt(0);
            __syncthreads();
            buf ^= 1;
        }
    } else
#endif
    {
        // ---- register-staged path (handles N-guarded tiles / fallback) ----
        uint4 ra[ACH], rb[BCH];
        auto loadTiles = [&](int kt) {
#pragma unroll
            for (int c = 0; c < ACH; ++c) {
                const int idx = tid + c * NT;
                if ((BM * 2) % NT == 0 || idx < BM * 2) {
                    const int row = idx >> 1, col = (idx & 1) * 16;
                    ra[c] = *(const uint4*)(A + (size_t)(m0 + row) * lda + kt * 32 + col);
                }
            }
#pragma unroll
            for (int c = 0; c < BCH; ++c) {
                const int idx = tid + c * NT;
                if ((BN * 2) % NT == 0 || idx < BN * 2) {
                    const int row = idx >> 1, col = (idx & 1) * 16;
                    const int nrow = n0 + row;
                    const int nsafe = (!NGUARD || nrow < N) ? nrow : 0;
                    uint4 t = *(const uint4*)(B + (size_t)nsafe * ldb + kt * 32 + col);
                    if (NGUARD) {
                        const bool ok = nrow < N;
                        t.x = ok ? t.x : 0u; t.y = ok ? t.y : 0u;
                        t.z = ok ? t.z : 0u; t.w = ok ? t.w : 0u;
                    }
                    rb[c] = t;
                }
            }
        };
        auto storeTiles = [&](int bufi) {
#pragma unroll
            for (int c = 0; c < ACH; ++c) {
                const int idx = tid + c * NT;
                if ((BM * 2) % NT == 0 || idx < BM * 2) {
                    const int row = idx >> 1, col = (idx & 1) * 16;
                    *(uint4*)&sA[bufi][row * LDT + col] = ra[c];
                }
            }
#pragma unroll
            for (int c = 0; c < BCH; ++c) {
                const int idx = tid + c * NT;
                if ((BN * 2) % NT == 0 || idx < BN * 2) {
                    const int row = idx >> 1, col = (idx & 1) * 16;
                    *(uint4*)&sB[bufi][row * LDT + col] = rb[c];
                }
            }
        };

        loadTiles(0);
        storeTiles(0);
        __syncthreads();
        for (int kt = 0; kt < nk; ++kt) {
            const bool has = (kt + 1) < nk;
            if (has) loadTiles(kt + 1);
            computeStep(buf);
            if (has) storeTiles(buf ^ 1);
            __syncthreads();
            buf ^= 1;
        }
    }

    // epilogue: C layout -> VGPR e: M = e + (lane<16 ? 0 : 8), N = lane&15
    const int crow = (lane >> 4) << 3;   // 0 or 8
    const int ccol = lane & 15;
#pragma unroll
    for (int j = 0; j < WNT; ++j) {
        const int n = n0 + wn + j * 16 + ccol;
        float bsv = 0.f;
        if (bias != nullptr && n < N) bsv = bias[n];
        if (n < n_limit) {
#pragma unroll
            for (int i = 0; i < WMT; ++i) {
                const int mbase = m0 + wm + i * 16 + crow;
#pragma unroll
                for (int e = 0; e < 8; ++e) {
                    const float val = acc[i][j][e] * alpha + bsv;
                    const size_t off = (size_t)(mbase + e) * ldc + n;
                    if (Cf) Cf[off] = val;
                    if (Cb) Cb[off] = (bf16)val;
                }
            }
        }
    }
}

// =====================================================================
// Prep kernels
// =====================================================================
__global__ void f32_to_bf16_kernel(const float* __restrict__ in,
                                   bf16* __restrict__ out, int n)
{
    const int i = blockIdx.x * blockDim.x + threadIdx.x;
    if (i < n) out[i] = (bf16)in[i];
}

// Wout [1000,1024] -> padded bf16 [1024,1024]; bout -> padded f32 [1024]
__global__ void pad_wout_kernel(const float* __restrict__ Wout,
                                const float* __restrict__ bout,
                                bf16* __restrict__ WoP,
                                float* __restrict__ boP)
{
    const int i = blockIdx.x * blockDim.x + threadIdx.x;  // 0 .. 1024*1024-1
    const int n = i >> 10;
    const int k = i & (MH - 1);
    WoP[i] = (n < MOUT) ? (bf16)Wout[(size_t)n * MH + k] : (bf16)0.f;
    if (i < MH) boP[i] = (i < MOUT) ? bout[i] : 0.f;
}

// oq[o,h] = sum_s operators[o,s]*Wq_op[h,s] + bq_op[h]   (tiny: 16x1024)
__global__ void oq_kernel(const float* __restrict__ operators,
                          const float* __restrict__ Wq,
                          const float* __restrict__ bq,
                          bf16* __restrict__ oqb)
{
    const int i = blockIdx.x * blockDim.x + threadIdx.x;  // 0..O*H-1
    if (i >= MO * MH) return;
    const int o = i >> 10;
    const int h = i & (MH - 1);
    const float* op = operators + (size_t)o * MS;
    const float* w  = Wq + (size_t)h * MS;
    float accv = bq[h];
#pragma unroll 8
    for (int s = 0; s < MS; ++s) accv += op[s] * w[s];
    oqb[i] = (bf16)accv;
}

// x[b,s,h] = word_emb[tokens[b,s]][h] + sinusoidal_pe(s,h); f32 + bf16
__global__ void embed_kernel(const int* __restrict__ tokens,
                             const float* __restrict__ word_emb,
                             float* __restrict__ x,
                             bf16* __restrict__ xb)
{
    const int i = blockIdx.x * blockDim.x + threadIdx.x;  // B*S*H
    const int h  = i & (MH - 1);
    const int bs = i >> 10;
    const int s  = bs & (MS - 1);
    const int tok = tokens[bs];
    const int j = h >> 1;
    // div = exp(-(2j)*ln(10000)/H)
    const float div = __expf((float)j * (-2.0f * 9.210340371976184f / (float)MH));
    const float ang = (float)s * div;
    const float pe  = (h & 1) ? __cosf(ang) : __sinf(ang);
    const float val = word_emb[(size_t)tok * MH + h] + pe;
    x[i]  = val;
    xb[i] = (bf16)val;
}

// =====================================================================
// Softmax / einsum kernels
// =====================================================================
// opw[b,s,:] = softmax over the O=16 logits of one row
__global__ void softmax_opw_kernel(const float* __restrict__ logits,
                                   float* __restrict__ opw)
{
    const int r = blockIdx.x * blockDim.x + threadIdx.x;  // 0..B*S-1
    if (r >= MROWS) return;
    const float* p = logits + (size_t)r * MO;
    float mx = -3.4e38f;
    float e[MO];
#pragma unroll
    for (int i = 0; i < MO; ++i) mx = fmaxf(mx, p[i]);
    float sum = 0.f;
#pragma unroll
    for (int i = 0; i < MO; ++i) { e[i] = __expf(p[i] - mx); sum += e[i]; }
    const float inv = 1.f / sum;
    float* q = opw + (size_t)r * MO;
#pragma unroll
    for (int i = 0; i < MO; ++i) q[i] = e[i] * inv;
}

// relw = softmax over s of logits[b,:,o]; fused:
// wmat[b,o,s] = relw[b,s,o] * operators[o,s]  (bf16)
// one wave32 per (b,o); each lane owns 16 strided s positions
__global__ void relw_wmat_kernel(const float* __restrict__ logits,
                                 const float* __restrict__ operators,
                                 bf16* __restrict__ wmat)
{
    const int gw = blockIdx.x * (blockDim.x >> 5) + (threadIdx.x >> 5);
    if (gw >= MB * MO) return;
    const int b = gw >> 4;
    const int o = gw & (MO - 1);
    const int lane = threadIdx.x & 31;

    const float* lg = logits + (size_t)b * MS * MO + o;   // stride MO over s
    float v[16];
    float mx = -3.4e38f;
#pragma unroll
    for (int i = 0; i < 16; ++i) {
        v[i] = lg[(size_t)(lane + i * 32) * MO];
        mx = fmaxf(mx, v[i]);
    }
#pragma unroll
    for (int off = 16; off > 0; off >>= 1)
        mx = fmaxf(mx, __shfl_xor(mx, off, 32));
    float sum = 0.f;
#pragma unroll
    for (int i = 0; i < 16; ++i) { v[i] = __expf(v[i] - mx); sum += v[i]; }
#pragma unroll
    for (int off = 16; off > 0; off >>= 1)
        sum += __shfl_xor(sum, off, 32);
    const float inv = 1.f / sum;

    const float* ops = operators + (size_t)o * MS;
    bf16* wp = wmat + ((size_t)b * MO + o) * MS;
#pragma unroll
    for (int i = 0; i < 16; ++i) {
        const int s = lane + i * 32;
        wp[s] = (bf16)(v[i] * inv * ops[s]);
    }
}

// op_out[b,o,h] = sum_s wmat[b,o,s] * v[b,s,h]
// grid: (H/256, O, B); block 256 threads (one h each)
__global__ void op_out_kernel(const bf16* __restrict__ wmat,
                              const bf16* __restrict__ v,
                              float* __restrict__ op_out)
{
    __shared__ float w[MS];
    const int b = blockIdx.z, o = blockIdx.y;
    const bf16* wp = wmat + ((size_t)b * MO + o) * MS;
    for (int i = threadIdx.x; i < MS; i += blockDim.x) w[i] = (float)wp[i];
    __syncthreads();
    const int h = blockIdx.x * blockDim.x + threadIdx.x;
    const bf16* vb = v + (size_t)b * MS * MH + h;
    float accv = 0.f;
#pragma unroll 4
    for (int s = 0; s < MS; ++s) accv += w[s] * (float)vb[(size_t)s * MH];
    op_out[((size_t)b * MO + o) * MH + h] = accv;
}

// out[b,s,h] = sum_o opw[b,s,o] * op_out[b,o,h];  x += out (f32 + bf16)
// grid: (H/256, S, B)
__global__ void out_residual_kernel(const float* __restrict__ opw,
                                    const float* __restrict__ op_out,
                                    float* __restrict__ x,
                                    bf16* __restrict__ xb)
{
    __shared__ float w[MO];
    const int b = blockIdx.z, s = blockIdx.y;
    if (threadIdx.x < MO)
        w[threadIdx.x] = opw[((size_t)b * MS + s) * MO + threadIdx.x];
    __syncthreads();
    const int h = blockIdx.x * blockDim.x + threadIdx.x;
    const float* oo = op_out + (size_t)b * MO * MH + h;
    float accv = 0.f;
#pragma unroll
    for (int o = 0; o < MO; ++o) accv += w[o] * oo[(size_t)o * MH];
    const size_t xi = ((size_t)b * MS + s) * MH + h;
    const float nv = x[xi] + accv;
    x[xi]  = nv;
    xb[xi] = (bf16)nv;
}

// =====================================================================
// Host launcher
// =====================================================================
extern "C" void kernel_launch(void* const* d_in, const int* in_sizes, int n_in,
                              void* d_out, int out_size, void* d_ws, size_t ws_size,
                              hipStream_t stream)
{
    (void)in_sizes; (void)n_in; (void)out_size; (void)ws_size;

    const int*   tokens    = (const int*)  d_in[0];
    const float* word_emb  = (const float*)d_in[1];
    const float* Wv        = (const float*)d_in[2];
    const float* bv        = (const float*)d_in[3];
    const float* Wk        = (const float*)d_in[4];
    const float* bk        = (const float*)d_in[5];
    const float* Wq_op     = (const float*)d_in[6];
    const float* bq_op     = (const float*)d_in[7];
    const float* operators = (const float*)d_in[8];
    const float* Wout      = (const float*)d_in[9];
    const float* bout      = (const float*)d_in[10];

    // workspace carve-out (all naturally aligned, power-of-two sizes)
    char* p = (char*)d_ws;
    float* x_f32 = (float*)p;  p += (size_t)MROWS * MH * 4;   // 64 MB
    bf16*  x_b   = (bf16*)p;   p += (size_t)MROWS * MH * 2;   // 32 MB
    bf16*  v_b   = (bf16*)p;   p += (size_t)MROWS * MH * 2;   // 32 MB
    bf16*  k_b   = (bf16*)p;   p += (size_t)MROWS * MH * 2;   // 32 MB
    bf16*  Wv_b  = (bf16*)p;   p += (size_t)MH * MH * 2;
    bf16*  Wk_b  = (bf16*)p;   p += (size_t)MH * MH * 2;
    bf16*  Wo_b  = (bf16*)p;   p += (size_t)MH * MH * 2;      // padded 1024x1024
    float* bo_p  = (float*)p;  p += (size_t)MH * 4;           // padded bias
    bf16*  oq_b  = (bf16*)p;   p += (size_t)MO * MH * 2;
    float* logits= (float*)p;  p += (size_t)MROWS * MO * 4;
    float* opw   = (float*)p;  p += (size_t)MROWS * MO * 4;
    bf16*  wmat  = (bf16*)p;   p += (size_t)MB * MO * MS * 2;
    float* opout = (float*)p;  p += (size_t)MB * MO * MH * 4;

    const float scale = 0.03125f;  // 1/sqrt(H) = 1/32

    // ---- prep ----
    f32_to_bf16_kernel<<<(MH * MH) / 256, 256, 0, stream>>>(Wv, Wv_b, MH * MH);
    f32_to_bf16_kernel<<<(MH * MH) / 256, 256, 0, stream>>>(Wk, Wk_b, MH * MH);
    pad_wout_kernel<<<(MH * MH) / 256, 256, 0, stream>>>(Wout, bout, Wo_b, bo_p);
    oq_kernel<<<(MO * MH) / 256, 256, 0, stream>>>(operators, Wq_op, bq_op, oq_b);
    embed_kernel<<<(MROWS * MH) / 256, 256, 0, stream>>>(tokens, word_emb, x_f32, x_b);

    const dim3 gemmBlk(128);
    const dim3 gridBig(MH / 128, MROWS / 128);   // 8 x 128 tiles
    const dim3 gridLog(1, MROWS / 128);          // N=16 -> single N tile

    // ---- layers (shared weights) ----
    for (int l = 0; l < MLAYERS; ++l) {
        // v = x @ Wv^T + bv      (bf16 out)
        gemm_bf16_wmma<128,128,4,4,false><<<gridBig, gemmBlk, 0, stream>>>(
            x_b, MH, Wv_b, MH, bv, nullptr, v_b, MH,
            MROWS, MH, MH, 1.0f, MH);
        // k = x @ Wk^T + bk      (bf16 out)
        gemm_bf16_wmma<128,128,4,4,false><<<gridBig, gemmBlk, 0, stream>>>(
            x_b, MH, Wk_b, MH, bk, nullptr, k_b, MH,
            MROWS, MH, MH, 1.0f, MH);
        // logits = (k @ oq^T) * scale   (f32 out, N=16, dedicated skinny tile)
        gemm_bf16_wmma<128,16,2,1,true><<<gridLog, gemmBlk, 0, stream>>>(
            k_b, MH, oq_b, MH, nullptr, logits, nullptr, MO,
            MROWS, MO, MH, scale, MO);
        // softmaxes
        softmax_opw_kernel<<<MROWS / 256, 256, 0, stream>>>(logits, opw);
        relw_wmat_kernel<<<(MB * MO) / 8, 256, 0, stream>>>(logits, operators, wmat);
        // op_out[b,o,h] = sum_s wmat * v
        op_out_kernel<<<dim3(MH / 256, MO, MB), 256, 0, stream>>>(wmat, v_b, opout);
        // x += sum_o opw * op_out
        out_residual_kernel<<<dim3(MH / 256, MS, MB), 256, 0, stream>>>(
            opw, opout, x_f32, x_b);
    }

    // ---- final projection: out = x @ Wout^T + bout  (N padded to 1024) ----
    gemm_bf16_wmma<128,128,4,4,false><<<gridBig, gemmBlk, 0, stream>>>(
        x_b, MH, Wo_b, MH, bo_p, (float*)d_out, nullptr, MOUT,
        MROWS, MH, MH, 1.0f, MOUT);
}
